// MultiModalMaskedAttention_78709570667317
// MI455X (gfx1250) — matrix operand order repaired and checked by reading, not compiled, not run on hardware
//
#include <hip/hip_runtime.h>
#include <hip/hip_bf16.h>

// ---------------------------------------------------------------------------
// MultiModalMaskedAttention for MI455X (gfx1250), wave32, WMMA f16->f32.
// B=16, M=N=DT=1024, HID=512.
// Wave tile 32x64 (8 WMMAs / 6 fragment loads per K-step), block tile 128x128.
// ---------------------------------------------------------------------------

#define B_   16
#define M_   1024
#define N_   1024
#define DT_  1024
#define HID_ 512
#define TT_  (B_ * M_)            // 16384 flattened tokens (same for oT)
#define INV_SQRT_ 0.03125f        // 1/sqrt(1024)
#define SQRT_DT_  32.0f
#define NEG_INF_  (-1.0e9f)

typedef __attribute__((ext_vector_type(16))) _Float16 v16h;
typedef __attribute__((ext_vector_type(8)))  _Float16 v8h;
typedef __attribute__((ext_vector_type(8)))  float    v8f;

union U16 { v16h v; v8h h[2]; };

// A fragment (16x32 f16, MxK): lane r (0..15) holds row r; halves = K
// {k0+8h .. k0+8h+7, k0+16+8h .. k0+16+8h+7}  (h = lane>>4)
__device__ inline v16h load_fragA(const _Float16* __restrict__ base, int row,
                                  int ld, int k, int h) {
  const _Float16* p = base + (size_t)row * ld + k + h * 8;
  U16 u;
  u.h[0] = *(const v8h*)(p);
  u.h[1] = *(const v8h*)(p + 16);
  return u.v;
}

// B fragment (32x16 f16, KxN): lane r holds column r; halves = K
// {k0+8h .. k0+8h+7, k0+16+8h .. k0+16+8h+7}  (h = lane>>4)
__device__ inline v16h load_fragB(const _Float16* __restrict__ base, int col,
                                  int ld, int k, int h) {
  const _Float16* p = base + (size_t)col * ld + k + h * 8;
  U16 u;
  u.h[0] = *(const v8h*)(p);
  u.h[1] = *(const v8h*)(p + 16);
  return u.v;
}

enum { EPI_F32 = 0, EPI_BIAS_F16 = 1, EPI_BIAS_F16T = 2, EPI_MASK_VT = 3, EPI_MASK_TV = 4 };

// Generic batched WMMA GEMM: C[row, col] = sum_k A[row,k] * B[col,k]  (+epilogue)
// Block = 256 threads = 8 waves (4 row x 2 col); wave tile 32x64; block 128x128.
// grid.x = cols/128, grid.y = rows/128, grid.z = batch.
template <int EPI>
__global__ __launch_bounds__(256) void gemm_wmma(
    const _Float16* __restrict__ A, long aBatch, int lda,
    const _Float16* __restrict__ Bm, long bBatch, int ldb,
    int K,
    float* __restrict__ Cf, _Float16* __restrict__ Ch, long cBatch, int ldc,
    const float* __restrict__ bias,
    const float* __restrict__ mask, long mBatch, int mld,
    float alpha) {
  const int lane = threadIdx.x & 31;
  const int wave = threadIdx.x >> 5;
  const int r = lane & 15;
  const int h = lane >> 4;
  const int row0 = blockIdx.y * 128 + (wave >> 1) * 32;
  const int col0 = blockIdx.x * 128 + (wave & 1) * 64;
  const int b = blockIdx.z;

  const _Float16* Ab = A  + (size_t)b * aBatch;
  const _Float16* Bb = Bm + (size_t)b * bBatch;

  v8f acc[2][4];
#pragma unroll
  for (int i = 0; i < 2; ++i)
#pragma unroll
    for (int j = 0; j < 4; ++j) acc[i][j] = (v8f)0.0f;

  for (int k = 0; k < K; k += 32) {
    v16h a0 = load_fragA(Ab, row0 + r,      lda, k, h);
    v16h a1 = load_fragA(Ab, row0 + 16 + r, lda, k, h);
    v16h b0 = load_fragB(Bb, col0 + r,      ldb, k, h);
    v16h b1 = load_fragB(Bb, col0 + 16 + r, ldb, k, h);
    v16h b2 = load_fragB(Bb, col0 + 32 + r, ldb, k, h);
    v16h b3 = load_fragB(Bb, col0 + 48 + r, ldb, k, h);
    acc[0][0] = __builtin_amdgcn_wmma_f32_16x16x32_f16(false, a0, false, b0, (short)0, acc[0][0], false, false);
    acc[0][1] = __builtin_amdgcn_wmma_f32_16x16x32_f16(false, a0, false, b1, (short)0, acc[0][1], false, false);
    acc[0][2] = __builtin_amdgcn_wmma_f32_16x16x32_f16(false, a0, false, b2, (short)0, acc[0][2], false, false);
    acc[0][3] = __builtin_amdgcn_wmma_f32_16x16x32_f16(false, a0, false, b3, (short)0, acc[0][3], false, false);
    acc[1][0] = __builtin_amdgcn_wmma_f32_16x16x32_f16(false, a1, false, b0, (short)0, acc[1][0], false, false);
    acc[1][1] = __builtin_amdgcn_wmma_f32_16x16x32_f16(false, a1, false, b1, (short)0, acc[1][1], false, false);
    acc[1][2] = __builtin_amdgcn_wmma_f32_16x16x32_f16(false, a1, false, b2, (short)0, acc[1][2], false, false);
    acc[1][3] = __builtin_amdgcn_wmma_f32_16x16x32_f16(false, a1, false, b3, (short)0, acc[1][3], false, false);
  }

  // Epilogue. C layout: VGPR g of lane -> element (row0+ti*16 + g + 8h, col0+tj*16 + r)
#pragma unroll
  for (int ti = 0; ti < 2; ++ti) {
#pragma unroll
    for (int tj = 0; tj < 4; ++tj) {
      const int mm = row0 + ti * 16 + 8 * h;   // first of 8 consecutive rows
      const int nn = col0 + tj * 16 + r;       // column
      if (EPI == EPI_F32) {
        float* Cb = Cf + (size_t)b * cBatch;
#pragma unroll
        for (int g = 0; g < 8; ++g)
          Cb[(size_t)(mm + g) * ldc + nn] = acc[ti][tj][g] * alpha;
      } else if (EPI == EPI_BIAS_F16) {
        const float bv = bias[nn];
#pragma unroll
        for (int g = 0; g < 8; ++g)
          Ch[(size_t)(mm + g) * ldc + nn] = (_Float16)(acc[ti][tj][g] + bv);
      } else if (EPI == EPI_BIAS_F16T) {
        // transposed, packed: output[nn, mm..mm+7] contiguous -> one b128 store
        const float bv = bias[nn];
        v8h pk;
#pragma unroll
        for (int g = 0; g < 8; ++g) pk[g] = (_Float16)(acc[ti][tj][g] + bv);
        *(v8h*)(Ch + (size_t)nn * ldc + mm) = pk;
      } else if (EPI == EPI_MASK_VT) {
        float* Cb = Cf + (size_t)b * cBatch;
        const float* mb = mask + (size_t)b * mBatch;
#pragma unroll
        for (int g = 0; g < 8; ++g) {
          float mv = mb[(size_t)(mm + g) * mld + nn];
          Cb[(size_t)(mm + g) * ldc + nn] =
              (mv != 0.0f) ? acc[ti][tj][g] * alpha : NEG_INF_;
        }
      } else if (EPI == EPI_MASK_TV) {
        float* Cb = Cf + (size_t)b * cBatch;
        const float* mb = mask + (size_t)b * mBatch;
#pragma unroll
        for (int g = 0; g < 8; ++g) {
          float mv = mb[(size_t)nn * mld + (mm + g)];   // transposed mask read
          Cb[(size_t)(mm + g) * ldc + nn] =
              (mv != 0.0f) ? acc[ti][tj][g] * alpha : NEG_INF_;
        }
      }
    }
  }
}

// ---------------------------------------------------------------------------
// Support kernels
// ---------------------------------------------------------------------------

__global__ void cvt_f32_f16(const float* __restrict__ src, _Float16* __restrict__ dst,
                            long n) {
  long i = (long)blockIdx.x * blockDim.x + threadIdx.x;
  long stride = (long)gridDim.x * blockDim.x;
  for (; i < n; i += stride) dst[i] = (_Float16)src[i];
}

// mean over tokens for oV and oT -> v_pool/t_pool [B, DT]
__global__ void pool_kernel(const float* __restrict__ oV, const float* __restrict__ oT,
                            float* __restrict__ v_pool, float* __restrict__ t_pool) {
  int t = blockIdx.x * blockDim.x + threadIdx.x;   // 0 .. 2*B*DT
  if (t >= 2 * B_ * DT_) return;
  int which = t / (B_ * DT_);
  int rem = t - which * (B_ * DT_);
  int b = rem / DT_, d = rem - b * DT_;
  const float* src = which ? oT : oV;
  float s = 0.0f;
  const float* p = src + (size_t)b * M_ * DT_ + d;
  for (int m = 0; m < M_; ++m) s += p[(size_t)m * DT_];
  (which ? t_pool : v_pool)[rem] = s * (1.0f / M_);
}

// tiny FFN: h = relu(W1 @ [vp;tp] + b1);  l = sigmoid(w2.h + b2);  logl[b] = log(l)
__global__ __launch_bounds__(256) void ffn_kernel(
    const float* __restrict__ v_pool, const float* __restrict__ t_pool,
    const float* __restrict__ w1, const float* __restrict__ b1,
    const float* __restrict__ w2, const float* __restrict__ b2,
    float* __restrict__ logl) {
  __shared__ float hbuf[HID_];
  int b = blockIdx.x;
  for (int j = threadIdx.x; j < HID_; j += blockDim.x) {
    const float* wr = w1 + (size_t)j * (2 * DT_);
    float s = b1[j];
    for (int i = 0; i < DT_; ++i) s += wr[i] * v_pool[b * DT_ + i];
    for (int i = 0; i < DT_; ++i) s += wr[DT_ + i] * t_pool[b * DT_ + i];
    hbuf[j] = s > 0.0f ? s : 0.0f;
  }
  __syncthreads();
  if (threadIdx.x == 0) {
    float s = b2[0];
    for (int j = 0; j < HID_; ++j) s += w2[j] * hbuf[j];
    float l = 1.0f / (1.0f + __expf(-s));
    logl[b] = __logf(l);
  }
}

// per-row threshold: T[row] = (LSE(sim*inv) + log l) * sqrt(dt); mask = sim >= T
__global__ __launch_bounds__(256) void rowstats_kernel(const float* __restrict__ sim,
                                                       const float* __restrict__ logl,
                                                       float* __restrict__ T) {
  __shared__ float red[256];
  int row = blockIdx.x;                 // global b*M + m
  const float* p = sim + (size_t)row * N_;
  float mx = -3.4e38f;
  for (int j = threadIdx.x; j < N_; j += 256) mx = fmaxf(mx, p[j]);
  red[threadIdx.x] = mx; __syncthreads();
  for (int s = 128; s > 0; s >>= 1) {
    if (threadIdx.x < s) red[threadIdx.x] = fmaxf(red[threadIdx.x], red[threadIdx.x + s]);
    __syncthreads();
  }
  mx = red[0]; __syncthreads();
  float se = 0.0f;
  for (int j = threadIdx.x; j < N_; j += 256) se += __expf((p[j] - mx) * INV_SQRT_);
  red[threadIdx.x] = se; __syncthreads();
  for (int s = 128; s > 0; s >>= 1) {
    if (threadIdx.x < s) red[threadIdx.x] += red[threadIdx.x + s];
    __syncthreads();
  }
  if (threadIdx.x == 0) {
    float lse = mx * INV_SQRT_ + __logf(red[0]);
    int b = row / M_;
    T[row] = (lse + logl[b]) * SQRT_DT_;
  }
}

__global__ void maskwrite_kernel(const float* __restrict__ sim,
                                 const float* __restrict__ T,
                                 float* __restrict__ maskOut, long n) {
  long i = (long)blockIdx.x * blockDim.x + threadIdx.x;
  long stride = (long)gridDim.x * blockDim.x;
  for (; i < n; i += stride) {
    long row = i / N_;
    maskOut[i] = (sim[i] >= T[row]) ? 1.0f : 0.0f;
  }
}

// row softmax over masked scores (f32, already scaled+masked) -> f16 probabilities
__global__ __launch_bounds__(256) void softmax_kernel(const float* __restrict__ S,
                                                      _Float16* __restrict__ P) {
  __shared__ float red[256];
  int row = blockIdx.x;
  const float* p = S + (size_t)row * N_;
  _Float16* o = P + (size_t)row * N_;
  float mx = -3.4e38f;
  for (int j = threadIdx.x; j < N_; j += 256) mx = fmaxf(mx, p[j]);
  red[threadIdx.x] = mx; __syncthreads();
  for (int s = 128; s > 0; s >>= 1) {
    if (threadIdx.x < s) red[threadIdx.x] = fmaxf(red[threadIdx.x], red[threadIdx.x + s]);
    __syncthreads();
  }
  mx = red[0]; __syncthreads();
  float se = 0.0f;
  for (int j = threadIdx.x; j < N_; j += 256) se += __expf(p[j] - mx);
  red[threadIdx.x] = se; __syncthreads();
  for (int s = 128; s > 0; s >>= 1) {
    if (threadIdx.x < s) red[threadIdx.x] += red[threadIdx.x + s];
    __syncthreads();
  }
  float rinv = 1.0f / red[0];
  for (int j = threadIdx.x; j < N_; j += 256)
    o[j] = (_Float16)(__expf(p[j] - mx) * rinv);
}

// ---------------------------------------------------------------------------
// Launch
// ---------------------------------------------------------------------------

extern "C" void kernel_launch(void* const* d_in, const int* in_sizes, int n_in,
                              void* d_out, int out_size, void* d_ws, size_t ws_size,
                              hipStream_t stream) {
  (void)in_sizes; (void)n_in; (void)out_size; (void)ws_size;

  const float* oV     = (const float*)d_in[0];
  const float* oT     = (const float*)d_in[1];
  const float* Wqv_w  = (const float*)d_in[2];
  const float* Wqv_b  = (const float*)d_in[3];
  const float* Wkt_w  = (const float*)d_in[4];
  const float* Wkt_b  = (const float*)d_in[5];
  const float* Wvt_w  = (const float*)d_in[6];
  const float* Wvt_b  = (const float*)d_in[7];
  const float* Wqt_w  = (const float*)d_in[8];
  const float* Wqt_b  = (const float*)d_in[9];
  const float* Wkv_w  = (const float*)d_in[10];
  const float* Wkv_b  = (const float*)d_in[11];
  const float* Wvv_w  = (const float*)d_in[12];
  const float* Wvv_b  = (const float*)d_in[13];
  const float* ffn1_w = (const float*)d_in[14];
  const float* ffn1_b = (const float*)d_in[15];
  const float* ffn2_w = (const float*)d_in[16];
  const float* ffn2_b = (const float*)d_in[17];

  const long OUTSZ = (long)B_ * M_ * N_;         // 16,777,216
  float* out_vt  = (float*)d_out;
  float* out_tv  = out_vt + OUTSZ;
  float* maskOut = out_vt + 2 * OUTSZ;

  // workspace layout (256B aligned slices)
  char* ws = (char*)d_ws;
  size_t off = 0;
  auto take = [&](size_t bytes) -> char* {
    char* p = ws + off;
    off = (off + bytes + 255) & ~(size_t)255;
    return p;
  };
  const size_t tokBytesH = (size_t)TT_ * DT_ * 2;     // 32 MB (f16 token matrix)
  _Float16* oV16 = (_Float16*)take(tokBytesH);
  _Float16* oT16 = (_Float16*)take(tokBytesH);
  _Float16* W16[6];
  for (int i = 0; i < 6; ++i) W16[i] = (_Float16*)take((size_t)DT_ * DT_ * 2);
  _Float16* Qv16 = (_Float16*)take(tokBytesH);
  _Float16* Kt16 = (_Float16*)take(tokBytesH);
  _Float16* Qt16 = (_Float16*)take(tokBytesH);
  _Float16* Kv16 = (_Float16*)take(tokBytesH);
  _Float16* VtT  = (_Float16*)take(tokBytesH);        // [DT, TT] transposed
  _Float16* VvT  = (_Float16*)take(tokBytesH);        // [DT, TT] transposed
  float* simBuf  = (float*)take((size_t)B_ * M_ * N_ * 4);  // 64 MB, reused as scores
  float* v_pool  = (float*)take((size_t)B_ * DT_ * 4);
  float* t_pool  = (float*)take((size_t)B_ * DT_ * 4);
  float* logl    = (float*)take((size_t)B_ * 4);
  float* Trow    = (float*)take((size_t)TT_ * 4);
  // P buffers reuse oV16/oT16 (dead after projections)
  _Float16* Pvt = oV16;
  _Float16* Ptv = oT16;

  const long TOK = (long)TT_ * DT_;

  // 1) f32 -> f16 converts
  cvt_f32_f16<<<4096, 256, 0, stream>>>(oV, oV16, TOK);
  cvt_f32_f16<<<4096, 256, 0, stream>>>(oT, oT16, TOK);
  const float* wsrc[6] = {Wqv_w, Wkt_w, Wvt_w, Wqt_w, Wkv_w, Wvv_w};
  for (int i = 0; i < 6; ++i)
    cvt_f32_f16<<<2048, 256, 0, stream>>>(wsrc[i], W16[i], (long)DT_ * DT_);

  // 2) pooled means + 3) gating FFN
  pool_kernel<<<(2 * B_ * DT_ + 255) / 256, 256, 0, stream>>>(oV, oT, v_pool, t_pool);
  ffn_kernel<<<B_, 256, 0, stream>>>(v_pool, t_pool, ffn1_w, ffn1_b, ffn2_w, ffn2_b, logl);

  // 4) sim = oV @ oT^T  (per batch), f32 out
  {
    dim3 g(N_ / 128, M_ / 128, B_);
    gemm_wmma<EPI_F32><<<g, 256, 0, stream>>>(
        oV16, (long)M_ * DT_, DT_,  oT16, (long)N_ * DT_, DT_,  DT_,
        simBuf, nullptr, (long)M_ * N_, N_,
        nullptr, nullptr, 0, 0, 1.0f);
  }

  // 5) row LSE -> threshold; 6) mask output
  rowstats_kernel<<<TT_, 256, 0, stream>>>(simBuf, logl, Trow);
  maskwrite_kernel<<<4096, 256, 0, stream>>>(simBuf, Trow, maskOut, OUTSZ);

  // 7) six projections: X @ W^T + b, f16 outputs (V's stored transposed [DT,TT])
  {
    dim3 g(DT_ / 128, TT_ / 128, 1);
    // Q_v = oV @ Wq_v^T
    gemm_wmma<EPI_BIAS_F16><<<g, 256, 0, stream>>>(
        oV16, 0, DT_, W16[0], 0, DT_, DT_,
        nullptr, Qv16, 0, DT_, Wqv_b, nullptr, 0, 0, 1.0f);
    // K_t = oT @ Wk_t^T
    gemm_wmma<EPI_BIAS_F16><<<g, 256, 0, stream>>>(
        oT16, 0, DT_, W16[1], 0, DT_, DT_,
        nullptr, Kt16, 0, DT_, Wkt_b, nullptr, 0, 0, 1.0f);
    // V_t = oT @ Wv_t^T  (transposed store)
    gemm_wmma<EPI_BIAS_F16T><<<g, 256, 0, stream>>>(
        oT16, 0, DT_, W16[2], 0, DT_, DT_,
        nullptr, VtT, 0, TT_, Wvt_b, nullptr, 0, 0, 1.0f);
    // Q_t = oT @ Wq_t^T
    gemm_wmma<EPI_BIAS_F16><<<g, 256, 0, stream>>>(
        oT16, 0, DT_, W16[3], 0, DT_, DT_,
        nullptr, Qt16, 0, DT_, Wqt_b, nullptr, 0, 0, 1.0f);
    // K_v = oV @ Wk_v^T
    gemm_wmma<EPI_BIAS_F16><<<g, 256, 0, stream>>>(
        oV16, 0, DT_, W16[4], 0, DT_, DT_,
        nullptr, Kv16, 0, DT_, Wkv_b, nullptr, 0, 0, 1.0f);
    // V_v = oV @ Wv_v^T  (transposed store)
    gemm_wmma<EPI_BIAS_F16T><<<g, 256, 0, stream>>>(
        oV16, 0, DT_, W16[5], 0, DT_, DT_,
        nullptr, VvT, 0, TT_, Wvv_b, nullptr, 0, 0, 1.0f);
  }

  // 8) V->T: masked scores, softmax, P @ V_t
  {
    dim3 g(N_ / 128, M_ / 128, B_);
    gemm_wmma<EPI_MASK_VT><<<g, 256, 0, stream>>>(
        Qv16, (long)M_ * DT_, DT_,  Kt16, (long)N_ * DT_, DT_,  DT_,
        simBuf, nullptr, (long)M_ * N_, N_,
        nullptr, maskOut, (long)M_ * N_, N_, INV_SQRT_);
  }
  softmax_kernel<<<TT_, 256, 0, stream>>>(simBuf, Pvt);
  {
    dim3 g(DT_ / 128, M_ / 128, B_);
    gemm_wmma<EPI_F32><<<g, 256, 0, stream>>>(
        Pvt, (long)M_ * N_, N_,  VtT, (long)N_, TT_,  N_,
        out_vt, nullptr, (long)M_ * DT_, DT_,
        nullptr, nullptr, 0, 0, 1.0f);
  }

  // 9) T->V: masked scores (transposed mask), softmax, P @ V_v
  {
    dim3 g(M_ / 128, N_ / 128, B_);
    gemm_wmma<EPI_MASK_TV><<<g, 256, 0, stream>>>(
        Qt16, (long)N_ * DT_, DT_,  Kv16, (long)M_ * DT_, DT_,  DT_,
        simBuf, nullptr, (long)N_ * M_, M_,
        nullptr, maskOut, (long)M_ * N_, N_, INV_SQRT_);
  }
  softmax_kernel<<<TT_, 256, 0, stream>>>(simBuf, Ptv);
  {
    dim3 g(DT_ / 128, N_ / 128, B_);
    gemm_wmma<EPI_F32><<<g, 256, 0, stream>>>(
        Ptv, (long)N_ * M_, M_,  VvT, (long)M_, TT_,  M_,
        out_tv, nullptr, (long)N_ * DT_, DT_,
        nullptr, nullptr, 0, 0, 1.0f);
  }
}
